// SparseLinear_74345883894235
// MI455X (gfx1250) — compile-verified, matrix operand order checked by reading, not run on hardware
//
#include <hip/hip_runtime.h>

// ---- CDNA5 vector types -----------------------------------------------------
typedef __attribute__((ext_vector_type(16))) __bf16 v16bf;
typedef __attribute__((ext_vector_type(8)))  __bf16 v8bf;
typedef __attribute__((ext_vector_type(8)))  float  v8f;
typedef __attribute__((ext_vector_type(4)))  unsigned int v4u;
typedef __attribute__((ext_vector_type(8)))  int v8i;
typedef __attribute__((ext_vector_type(4)))  int v4i;

// ---- problem dims (from reference) ------------------------------------------
#define B_  8
#define S_  2048
#define K_  4096
#define M_  4096

// ---- tiling ------------------------------------------------------------------
#define BM 128      // C tile rows (M)
#define BN 128      // C tile cols (S)
#define KT 64       // K tile
#define THREADS 256 // 8 waves of 32

// ---- feature detect: Tensor Data Mover builtins ------------------------------
#if defined(__has_builtin)
#  if __has_builtin(__builtin_amdgcn_tensor_load_to_lds) && \
      __has_builtin(__builtin_amdgcn_s_wait_tensorcnt)
#    define USE_TDM 1
#  endif
#endif

// f32 -> bf16 round-to-nearest-even (bit pattern)
static __device__ __forceinline__ unsigned short f2bf(float f) {
    unsigned int u = __float_as_uint(f);
    unsigned int r = u + 0x7FFFu + ((u >> 16) & 1u);
    return (unsigned short)(r >> 16);
}

// ---- pack W * mask -> bf16 ---------------------------------------------------
__global__ __launch_bounds__(256)
void pack_w_kernel(const float* __restrict__ w,
                   const unsigned char* __restrict__ mask,
                   unsigned short* __restrict__ wbf) {
    size_t i = ((size_t)blockIdx.x * 256 + threadIdx.x) * 4;
    float4 v = *(const float4*)(w + i);
    uchar4 m = *(const uchar4*)(mask + i);
    ushort4 o;
    o.x = f2bf(m.x ? v.x : 0.0f);
    o.y = f2bf(m.y ? v.y : 0.0f);
    o.z = f2bf(m.z ? v.z : 0.0f);
    o.w = f2bf(m.w ? v.w : 0.0f);
    *(ushort4*)(wbf + i) = o;
}

// ---- pack x -> bf16 ----------------------------------------------------------
__global__ __launch_bounds__(256)
void pack_x_kernel(const float* __restrict__ x,
                   unsigned short* __restrict__ xbf) {
    size_t i = ((size_t)blockIdx.x * 256 + threadIdx.x) * 4;
    float4 v = *(const float4*)(x + i);
    ushort4 o;
    o.x = f2bf(v.x);
    o.y = f2bf(v.y);
    o.z = f2bf(v.z);
    o.w = f2bf(v.w);
    *(ushort4*)(xbf + i) = o;
}

#if USE_TDM
// Issue one TDM load: 2D tile [tile_dim1=128 rows][tile_dim0=64 halfs] from a
// row-major bf16 tensor with row stride K_, packed contiguously into LDS.
static __device__ __forceinline__ void tdm_load_tile(const unsigned short* gsrc,
                                                     unsigned int lds_off,
                                                     unsigned int nrows_tensor) {
    unsigned long long ga = (unsigned long long)(uintptr_t)gsrc;
    v4u g0;
    g0[0] = 1u;                          // count=1 (valid user descriptor)
    g0[1] = lds_off;                     // lds_addr (bytes)
    g0[2] = (unsigned int)ga;            // global_addr[31:0]
    g0[3] = (unsigned int)((ga >> 32) & 0x01FFFFFFu) | (2u << 30);  // [56:32] | type=2
    v8i g1;
    g1[0] = (int)(1u << 16);             // workgroup_mask=0, data_size=1 (2 bytes)
    g1[1] = (int)((unsigned)(K_ & 0xFFFF) << 16);           // tensor_dim0[15:0]=K
    g1[2] = (int)((nrows_tensor & 0xFFFFu) << 16);          // tensor_dim1[15:0]
    g1[3] = (int)((unsigned)KT << 16);   // tile_dim0 = 64
    g1[4] = (int)BM;                     // tile_dim1 = 128 rows
    g1[5] = (int)K_;                     // tensor_dim0_stride = K
    g1[6] = 0;
    g1[7] = 0;
    v4i z4 = {0, 0, 0, 0};
#if __clang_major__ >= 23
    v8i z8 = {0, 0, 0, 0, 0, 0, 0, 0};
    __builtin_amdgcn_tensor_load_to_lds(g0, g1, z4, z4, z8, 0);   // 6-arg form
#else
    __builtin_amdgcn_tensor_load_to_lds(g0, g1, z4, z4, 0);       // 5-arg form
#endif
}
#endif

// ---- WMMA GEMM: out[b,m,s] = sum_k Wbf[m,k] * Xbf[b,s,k] ---------------------
__global__ __launch_bounds__(THREADS)
void spmm_wmma_kernel(const unsigned short* __restrict__ Wbf,   // [M,K] bf16
                      const unsigned short* __restrict__ Xbf,   // [B,S,K] bf16
                      float* __restrict__ out) {                // [B,M,S]
    __shared__ __align__(16) unsigned short sA[2][BM * KT];  // W tile, 16 KB each
    __shared__ __align__(16) unsigned short sB[2][BN * KT];  // X tile, 16 KB each

    const int tid  = threadIdx.x;
    const int lane = tid & 31;
    const int wid  = tid >> 5;       // 0..7
    const int wm   = wid & 3;        // 4 waves across M (4*32 = 128)
    const int wn   = wid >> 2;       // 2 waves across S (2*64 = 128)
    const int hi   = lane >> 4;      // half-wave select (ISA K-chunk select)
    const int l16  = lane & 15;      // row/col within 16x16 tile

    const int bm = blockIdx.x * BM;
    const int bs = blockIdx.y * BN;
    const int b  = blockIdx.z;

    const unsigned short* Xb = Xbf + (size_t)b * S_ * K_;

    v8f acc[2][4];
    const v8f vzero = {0.f, 0.f, 0.f, 0.f, 0.f, 0.f, 0.f, 0.f};
#pragma unroll
    for (int mt = 0; mt < 2; ++mt)
#pragma unroll
        for (int nt = 0; nt < 4; ++nt) acc[mt][nt] = vzero;

#if USE_TDM
    // Workgroup-relative LDS byte offsets (generic addr low 32 bits = LDS offset)
    const unsigned int offA0 = (unsigned int)(uintptr_t)&sA[0][0];
    const unsigned int offA1 = (unsigned int)(uintptr_t)&sA[1][0];
    const unsigned int offB0 = (unsigned int)(uintptr_t)&sB[0][0];
    const unsigned int offB1 = (unsigned int)(uintptr_t)&sB[1][0];

    auto load_tile_tdm = [&](int buf, int k0) {
        tdm_load_tile(Wbf + (size_t)bm * K_ + k0, buf ? offA1 : offA0, M_);
        tdm_load_tile(Xb  + (size_t)bs * K_ + k0, buf ? offB1 : offB0, S_);
    };
#else
    // Fallback loader: 256 threads x 16B = 4 KB/pass; 16 KB tile -> 4 passes
    const int lrow = tid >> 3;        // 0..31
    const int lcol = (tid & 7) * 8;   // 0..56 step 8 halfs
    auto load_tile = [&](int buf, int k0) {
#pragma unroll
        for (int p = 0; p < 4; ++p) {
            int row = p * 32 + lrow;
            uint4 va = *(const uint4*)(Wbf + (size_t)(bm + row) * K_ + k0 + lcol);
            uint4 vb = *(const uint4*)(Xb  + (size_t)(bs + row) * K_ + k0 + lcol);
            *(uint4*)&sA[buf][row * KT + lcol] = va;
            *(uint4*)&sB[buf][row * KT + lcol] = vb;
        }
    };
#endif

    // Assemble one 16-bf16 fragment for this lane.
    // ISA 16-bit A 16x32 layout: half-wave 0 takes K {0..7, 16..23},
    // half-wave 1 takes K {8..15, 24..31} -> two contiguous 8-half chunks.
    auto frag16 = [&](const unsigned short* base, int row, int kk) -> v16bf {
        const __bf16* p = (const __bf16*)(base + row * KT + kk);
        v8bf lo = *(const v8bf*)(p + hi * 8);         // ds_load_b128
        v8bf up = *(const v8bf*)(p + 16 + hi * 8);    // ds_load_b128
        return __builtin_shufflevector(lo, up, 0, 1, 2, 3, 4, 5, 6, 7,
                                               8, 9, 10, 11, 12, 13, 14, 15);
    };

    // ---- prologue: fill buffer 0 --------------------------------------------
#if USE_TDM
    if (wid == 0) {
        load_tile_tdm(0, 0);
        __builtin_amdgcn_s_wait_tensorcnt(0);
    }
#else
    load_tile(0, 0);
#endif
    __syncthreads();

    const int NK = K_ / KT;  // 64
    for (int kt = 0; kt < NK; ++kt) {
        const int cur = kt & 1;

#if USE_TDM
        if (wid == 0 && kt + 1 < NK) load_tile_tdm(cur ^ 1, (kt + 1) * KT);
#else
        if (kt + 2 < NK) {  // speculative L2 prefetch two tiles ahead
            __builtin_prefetch(Wbf + (size_t)(bm + lrow) * K_ + (kt + 2) * KT + lcol, 0, 0);
            __builtin_prefetch(Xb  + (size_t)(bs + lrow) * K_ + (kt + 2) * KT + lcol, 0, 0);
        }
        if (kt + 1 < NK) load_tile(cur ^ 1, (kt + 1) * KT);
#endif

#pragma unroll
        for (int kk = 0; kk < KT; kk += 32) {
            v16bf af[2], bfr[4];
#pragma unroll
            for (int mt = 0; mt < 2; ++mt)
                af[mt] = frag16(sA[cur], wm * 32 + mt * 16 + l16, kk);
#pragma unroll
            for (int nt = 0; nt < 4; ++nt)
                bfr[nt] = frag16(sB[cur], wn * 64 + nt * 16 + l16, kk);
#pragma unroll
            for (int mt = 0; mt < 2; ++mt)
#pragma unroll
                for (int nt = 0; nt < 4; ++nt)
                    acc[mt][nt] = __builtin_amdgcn_wmma_f32_16x16x32_bf16(
                        false, af[mt], false, bfr[nt],
                        (short)0, acc[mt][nt], false, false);
        }

#if USE_TDM
        if (wid == 0) __builtin_amdgcn_s_wait_tensorcnt(0);
#endif
        __syncthreads();
    }

    // C/D layout: VGPR r holds M = r + 8*hi, lane&15 = N
#pragma unroll
    for (int mt = 0; mt < 2; ++mt) {
#pragma unroll
        for (int nt = 0; nt < 4; ++nt) {
            const int gs = bs + wn * 64 + nt * 16 + l16;
#pragma unroll
            for (int r = 0; r < 8; ++r) {
                const int gm = bm + wm * 32 + mt * 16 + r + hi * 8;
                out[((size_t)b * M_ + gm) * S_ + gs] = acc[mt][nt][r];
            }
        }
    }
}

extern "C" void kernel_launch(void* const* d_in, const int* in_sizes, int n_in,
                              void* d_out, int out_size, void* d_ws, size_t ws_size,
                              hipStream_t stream) {
    const float*         x    = (const float*)d_in[0];          // [B,S,K]
    const float*         w    = (const float*)d_in[1];          // [M,K]
    const unsigned char* mask = (const unsigned char*)d_in[2];  // [M,K] bool
    float*               out  = (float*)d_out;                  // [B,M,S]

    unsigned short* wbf = (unsigned short*)d_ws;                // 32 MB
    unsigned short* xbf = wbf + (size_t)M_ * K_;                // 128 MB

    {   // pack masked weight -> bf16
        unsigned int nblk = (unsigned int)((size_t)M_ * K_ / 4 / 256);  // 16384
        pack_w_kernel<<<dim3(nblk), 256, 0, stream>>>(w, mask, wbf);
    }
    {   // pack activations -> bf16
        unsigned int nblk = (unsigned int)((size_t)B_ * S_ * K_ / 4 / 256);  // 65536
        pack_x_kernel<<<dim3(nblk), 256, 0, stream>>>(x, xbf);
    }

    dim3 grid(M_ / BM, S_ / BN, B_);  // 32 x 16 x 8 = 4096 workgroups
    spmm_wmma_kernel<<<grid, THREADS, 0, stream>>>(wbf, xbf, out);
}
